// swinu2net3plus_61701500174568
// MI455X (gfx1250) — matrix operand order repaired and checked by reading, not compiled, not run on hardware
//
#include <hip/hip_runtime.h>
#include <hip/hip_bf16.h>
#include <math.h>

typedef _Float16 f16;
typedef __attribute__((ext_vector_type(8)))  _Float16 v8h;
typedef __attribute__((ext_vector_type(16))) _Float16 v16h;
typedef __attribute__((ext_vector_type(8)))  float    v8f;
typedef __attribute__((ext_vector_type(4)))  float    v4f;

#define WMMA_F16(a,b,c) __builtin_amdgcn_wmma_f32_16x16x32_f16(false,(a),false,(b),(short)0,(c),false,false)

// ---------------- fragment helpers (CDNA5 WMMA VGPR layouts, CDNA5 ISA) ----------------
__device__ inline v8f zf8(){
  v8f z;
  #pragma unroll
  for (int i=0;i<8;++i) z[i]=0.f;
  return z;
}
__device__ inline v8h zh8(){
  v8h z;
  #pragma unroll
  for (int i=0;i<8;++i) z[i]=(f16)0.f;
  return z;
}

__device__ inline v16h mk16(v8h lo, v8h hi){
  union { v16h v; v8h h[2]; } u; u.h[0]=lo; u.h[1]=hi; return u.v;
}

// A matrix 16x32 f16: lane<16 -> row=lane, K halves {0..7,16..23}; lane>=16 -> row=lane-16, K {8..15,24..31}
__device__ inline v16h load_a_frag(const f16* __restrict__ A, int row0, int k0, int ld){
  int lane = threadIdx.x & 31;
  int hi = lane >> 4;
  const f16* p = A + (size_t)(row0 + (lane & 15)) * ld + k0 + hi*8;
  return mk16(*(const v8h*)p, *(const v8h*)(p + 16));
}
// B matrix 32x16 f16 (stored [N,K] row-major): lane<16 col=lane K {0..7,16..23}; lane>=16 col=lane-16 K {8..15,24..31}
__device__ inline v16h load_b_frag(const f16* __restrict__ Bw, int col0, int k0, int ld){
  int lane = threadIdx.x & 31;
  const f16* p = Bw + (size_t)(col0 + (lane & 15)) * ld + k0 + (lane >> 4)*8;
  return mk16(*(const v8h*)p, *(const v8h*)(p + 16));
}

// ---------------- generic WMMA GEMM: out(m,n) = sum_k A[m,k]*W[n,k], epilogue fused ------
// Wave tile: 64 (M) x 64 (N); 16 WMMA per K-step vs 8 fragment loads (B reused over 4 M-tiles).
template<int KK, class Epi>
__global__ __launch_bounds__(128) void k_gemm(const f16* __restrict__ A,
                                              const f16* __restrict__ Bw, Epi epi){
  int nbase = blockIdx.x * 64;
  int mbase = (blockIdx.y * blockDim.y + threadIdx.y) * 64;
  int lane = threadIdx.x & 31;
  v8f acc[4][4];
  #pragma unroll
  for (int mt=0; mt<4; ++mt)
    #pragma unroll
    for (int nt=0; nt<4; ++nt) acc[mt][nt] = zf8();
  #pragma unroll
  for (int k0 = 0; k0 < KK; k0 += 32){
    v16h b[4];
    #pragma unroll
    for (int nt=0; nt<4; ++nt) b[nt] = load_b_frag(Bw, nbase + nt*16, k0, KK);
    #pragma unroll
    for (int mt=0; mt<4; ++mt){
      v16h a = load_a_frag(A, mbase + mt*16, k0, KK);
      #pragma unroll
      for (int nt=0; nt<4; ++nt)
        acc[mt][nt] = WMMA_F16(a, b[nt], acc[mt][nt]);
    }
  }
  int hi = lane >> 4, lc = lane & 15;
  #pragma unroll
  for (int mt=0; mt<4; ++mt)
    #pragma unroll
    for (int nt=0; nt<4; ++nt)
      #pragma unroll
      for (int r=0; r<8; ++r)
        epi(mbase + mt*16 + r + hi*8, nbase + nt*16 + lc, acc[mt][nt][r]);
}

// ---------------- epilogues ----------------
struct EpiQKV {  // q: [widx][head][tok][hd] *0.25 ; k: same ; v: TRANSPOSED [widx][head][hd][tok]
  const float* bias; f16* out;
  __device__ void operator()(int m, int n, float v) const {
    v += bias[n];
    int which = n >> 7, hn = n & 127, head = hn >> 4, d = hn & 15;
    if (which == 0) v *= 0.25f;                  // HD^-0.5 = 16^-0.5
    int widx = m >> 6, tok = m & 63;
    size_t base = ((size_t)(which*1024 + widx)*8 + head)*1024;
    size_t off  = (which == 2) ? (size_t)(d*64 + tok) : (size_t)(tok*16 + d);
    out[base + off] = (f16)v;
  }
};
struct EpiProjRes {       // window reverse + roll(+shift) + residual add
  const float* bias; float* h; int shift;
  __device__ void operator()(int m, int n, float v) const {
    v += bias[n];
    int widx = m >> 6, tok = m & 63;
    int b = widx >> 9, w = widx & 511;
    int gh = ((w >> 6)       * 4 + (tok >> 4)       + shift) & 31;
    int gw = (((w >> 3) & 7) * 4 + ((tok >> 2) & 3) + shift) & 31;
    int gt = ((w & 7)        * 4 + (tok & 3)        + shift) & 31;
    size_t t = (size_t)((b * 32 + gh) * 32 + gw) * 32 + gt;
    h[t * 128 + n] += v;
  }
};
struct EpiFC1Gelu {
  const float* bias; f16* out;
  __device__ void operator()(int m, int n, float v) const {
    v += bias[n];
    v = 0.5f * v * (1.f + erff(v * 0.70710678f));
    out[(size_t)m * 512 + n] = (f16)v;
  }
};
struct EpiFC2Res {
  const float* bias; float* h;
  __device__ void operator()(int m, int n, float v) const {
    h[(size_t)m * 128 + n] += v + bias[n];
  }
};
struct EpiPW {            // pointwise conv + BN(eval) + ReLU -> NCDHW
  const float* bias; const float* bg; const float* bb; float* out;
  __device__ void operator()(int m, int n, float v) const {
    v += bias[n];
    v = v * 0.9999950000375f * bg[n] + bb[n];    // 1/sqrt(1+1e-5)
    out[((size_t)((m >> 15) * 256 + n) << 15) | (size_t)(m & 32767)] = fmaxf(v, 0.f);
  }
};

// ---------------- LayerNorm (+ optional shift-roll + window partition) ----------------
__global__ __launch_bounds__(32) void k_ln(const float* __restrict__ h,
                                           const float* __restrict__ g,
                                           const float* __restrict__ be,
                                           f16* __restrict__ out, int shift, int winorder){
  int tok = blockIdx.x;
  int src;
  if (winorder){
    int widx = tok >> 6, t = tok & 63;
    int b = widx >> 9, w = widx & 511;
    int gh = ((w >> 6)       * 4 + (t >> 4)       + shift) & 31;
    int gw = (((w >> 3) & 7) * 4 + ((t >> 2) & 3) + shift) & 31;
    int gt = ((w & 7)        * 4 + (t & 3)        + shift) & 31;
    src = ((b * 32 + gh) * 32 + gw) * 32 + gt;
  } else src = tok;
  const float* p = h + (size_t)src * 128;
  int lane = threadIdx.x;
  float x[4]; float s = 0.f, s2 = 0.f;
  #pragma unroll
  for (int j=0;j<4;++j){ x[j] = p[lane*4+j]; s += x[j]; s2 += x[j]*x[j]; }
  #pragma unroll
  for (int msk=16; msk>=1; msk>>=1){ s += __shfl_xor(s, msk); s2 += __shfl_xor(s2, msk); }
  float mean = s * (1.f/128.f);
  float var  = s2 * (1.f/128.f) - mean*mean;
  float rstd = rsqrtf(var + 1e-5f);
  f16* o = out + (size_t)tok * 128 + lane*4;
  #pragma unroll
  for (int j=0;j<4;++j)
    o[j] = (f16)((x[j]-mean)*rstd*g[lane*4+j] + be[lane*4+j]);
}

// ---- relative position bias table, lane-vector layout: btab[head][n][m&15][m>>4] ----
__global__ void k_btab(const float* __restrict__ rpb, float* __restrict__ btab){
  int idx = blockIdx.x*256 + threadIdx.x;          // 8*64*64 = 32768
  int hd = idx >> 12, n = (idx >> 6) & 63, m = idx & 63;
  int d0 = (n >> 4)       - (m >> 4)       + 3;
  int d1 = ((n >> 2) & 3) - ((m >> 2) & 3) + 3;
  int d2 = (n & 3)        - (m & 3)        + 3;
  int ri = (d0 * 7 + d1) * 7 + d2;
  btab[(size_t)(hd*64 + n)*64 + (m & 15)*4 + (m >> 4)] = rpb[ri * 8 + hd];
}

__device__ inline int regcnt(int gh, int gw, int gt){
  int rh = (gh < 28) ? 0 : ((gh < 30) ? 1 : 2);
  int rw = (gw < 28) ? 0 : ((gw < 30) ? 1 : 2);
  int rt = (gt < 28) ? 0 : ((gt < 30) ? 1 : 2);
  return 9*rh + 3*rw + rt;
}

// ---------------- attention: one wave per (window, head) ----------------
__global__ __launch_bounds__(32) void k_attn(const f16* __restrict__ qkv,
                                             const float* __restrict__ btab,
                                             f16* __restrict__ out, int shift){
  __shared__ __align__(16) f16 P[16 * 64];
  int wg = blockIdx.x;
  int widx = wg >> 3, head = wg & 7;
  const f16* q  = qkv + (size_t)(widx        * 8 + head) * 1024;   // [tok][hd]
  const f16* kk = qkv + (size_t)((1024+widx) * 8 + head) * 1024;   // [tok][hd]
  const f16* vv = qkv + (size_t)((2048+widx) * 8 + head) * 1024;   // [hd][tok] (transposed)
  int lane = threadIdx.x, hi = lane >> 4, lc = lane & 15;
  int w = widx & 511;
  int wh = (w >> 6) * 4, ww = ((w >> 3) & 7) * 4, wt = (w & 7) * 4;
  int cntc[4];
  #pragma unroll
  for (int nt=0; nt<4; ++nt){
    int tc = nt*16 + lc;
    cntc[nt] = regcnt(wh + (tc >> 4), ww + ((tc >> 2) & 3), wt + (tc & 3));
  }
  for (int mt=0; mt<4; ++mt){
    // S = q @ k^T  (K=16 padded to 32 with zeroed upper fragment halves)
    v8h z8 = zh8();
    int qrow = mt*16 + lc;
    v16h a = mk16(*(const v8h*)(q + qrow*16 + hi*8), z8);
    v8f s[4];
    #pragma unroll
    for (int nt=0; nt<4; ++nt){
      v16h b;
      const f16* kp = kk + (nt*16 + lc) * 16 + hi*8;
      b = mk16(*(const v8h*)kp, z8);
      s[nt] = WMMA_F16(a, b, zf8());
    }
    // bias + shift mask + softmax per row (16-lane half-wave reductions)
    #pragma unroll
    for (int r=0; r<8; ++r){
      int row = mt*16 + r + hi*8;
      int cntr = regcnt(wh + (row >> 4), ww + ((row >> 2) & 3), wt + (row & 3));
      v4f bv = *(const v4f*)(btab + (size_t)(head*64 + row)*64 + lc*4); // 4 nt biases, 1 b128
      float mx = -1e30f;
      #pragma unroll
      for (int nt=0; nt<4; ++nt){
        float xv = s[nt][r] + bv[nt];
        if (shift && cntc[nt] != cntr) xv -= 100.f;
        s[nt][r] = xv;
        mx = fmaxf(mx, xv);
      }
      #pragma unroll
      for (int msk=8; msk>=1; msk>>=1) mx = fmaxf(mx, __shfl_xor(mx, msk));
      float sum = 0.f;
      #pragma unroll
      for (int nt=0; nt<4; ++nt){ float e = expf(s[nt][r] - mx); s[nt][r] = e; sum += e; }
      #pragma unroll
      for (int msk=8; msk>=1; msk>>=1) sum += __shfl_xor(sum, msk);
      float inv = 1.f / sum;
      int rr = r + hi*8;
      #pragma unroll
      for (int nt=0; nt<4; ++nt) P[rr*64 + nt*16 + lc] = (f16)(s[nt][r] * inv);
    }
    __syncthreads();
    // O = P @ V   (16x16 result, K=64 -> two WMMA); V transposed -> contiguous B frags
    v8f o = zf8();
    #pragma unroll
    for (int k0=0; k0<64; k0+=32){
      const f16* pp = P + lc*64 + k0 + hi*8;
      v16h a2 = mk16(*(const v8h*)pp, *(const v8h*)(pp + 16));
      const f16* vp = vv + lc*64 + k0 + hi*8;    // col=hd (lane), K=key tokens contiguous
      v16h b2 = mk16(*(const v8h*)vp, *(const v8h*)(vp + 16));
      o = WMMA_F16(a2, b2, o);
    }
    __syncthreads();
    #pragma unroll
    for (int r=0; r<8; ++r){
      int row = mt*16 + r + hi*8;
      out[((size_t)(widx*64 + row))*128 + head*16 + lc] = (f16)o[r];
    }
  }
}

// ---------------- layout / conversion / depthwise ----------------
__global__ void k_tr(const float* __restrict__ x, float* __restrict__ h){
  int idx = blockIdx.x*256 + threadIdx.x;          // B*C*32768 = 8388608
  int sp = idx & 32767, bc = idx >> 15, c = bc & 127, b = bc >> 7;
  h[((size_t)(b*32768 + sp))*128 + c] = x[idx];
}
__global__ void k_cvt(const float* __restrict__ s, f16* __restrict__ d, int n){
  int i = blockIdx.x*256 + threadIdx.x;
  if (i < n) d[i] = (f16)s[i];
}
__global__ void k_dw(const float* __restrict__ h, const float* __restrict__ w,
                     f16* __restrict__ out){
  int idx = blockIdx.x*256 + threadIdx.x;          // token*128 + c
  int c = idx & 127, tokn = idx >> 7;
  int b = tokn >> 15, sp = tokn & 32767;
  int hh = sp >> 10, w2 = (sp >> 5) & 31, tt = sp & 31;
  float acc = 0.f;
  #pragma unroll
  for (int kd=0; kd<3; ++kd){
    int ih = hh + kd - 1; if (ih < 0 || ih > 31) continue;
    #pragma unroll
    for (int kh=0; kh<3; ++kh){
      int iw = w2 + kh - 1; if (iw < 0 || iw > 31) continue;
      #pragma unroll
      for (int kw=0; kw<3; ++kw){
        int it = tt + kw - 1; if (it < 0 || it > 31) continue;
        acc += h[((size_t)((b << 15) + (ih << 10) + (iw << 5) + it))*128 + c]
             * w[c*27 + kd*9 + kh*3 + kw];
      }
    }
  }
  out[idx] = (f16)acc;
}

// ---------------- host launch ----------------
extern "C" void kernel_launch(void* const* d_in, const int* in_sizes, int n_in,
                              void* d_out, int out_size, void* d_ws, size_t ws_size,
                              hipStream_t stream){
  const float* x     = (const float*)d_in[0];
  const float* ln1_g = (const float*)d_in[1];
  const float* ln1_b = (const float*)d_in[2];
  const float* qkv_w = (const float*)d_in[3];
  const float* qkv_b = (const float*)d_in[4];
  const float* proj_w= (const float*)d_in[5];
  const float* proj_b= (const float*)d_in[6];
  const float* rpb   = (const float*)d_in[7];
  const float* ln2_g = (const float*)d_in[8];
  const float* ln2_b = (const float*)d_in[9];
  const float* fc1_w = (const float*)d_in[10];
  const float* fc1_b = (const float*)d_in[11];
  const float* fc2_w = (const float*)d_in[12];
  const float* fc2_b = (const float*)d_in[13];
  const float* dw_w  = (const float*)d_in[14];
  const float* pw_w  = (const float*)d_in[15];
  const float* pw_b  = (const float*)d_in[16];
  const float* bn_g  = (const float*)d_in[17];
  const float* bn_b  = (const float*)d_in[18];
  float* out = (float*)d_out;

  char* wsb = (char*)d_ws;
  float* hbuf  = (float*)wsb;                              // 65536*128 f32   (33.5 MB)
  f16*   xb    = (f16*)(wsb + 33554432);                   // 65536*128 f16   (16.8 MB)
  f16*   attno = (f16*)(wsb + 50331648);                   // 65536*128 f16   (16.8 MB)
  f16*   big   = (f16*)(wsb + 67108864);                   // qkv then z      (67.1 MB)
  char*  wf    = wsb + 134217728;
  f16* qkvw = (f16*)wf;                                    // 4*384*128
  f16* projw = qkvw + 196608;                              // 4*128*128
  f16* fc1w  = projw + 65536;                              // 4*512*128
  f16* fc2w  = fc1w + 262144;                              // 4*128*512
  f16* pww   = fc2w + 262144;                              // 256*128
  float* btab = (float*)(wf + 2*819200);                   // 8*64*64 f32

  k_tr<<<32768, 256, 0, stream>>>(x, hbuf);
  k_cvt<<<196608/256, 256, 0, stream>>>(qkv_w, qkvw, 196608);
  k_cvt<<<65536/256,  256, 0, stream>>>(proj_w, projw, 65536);
  k_cvt<<<262144/256, 256, 0, stream>>>(fc1_w, fc1w, 262144);
  k_cvt<<<262144/256, 256, 0, stream>>>(fc2_w, fc2w, 262144);
  k_cvt<<<32768/256,  256, 0, stream>>>(pw_w, pww, 32768);

  dim3 blk(32, 4);                                        // 4 waves, 64 M-rows each
  for (int i = 0; i < 4; ++i){
    int shift = (i & 1) ? 2 : 0;
    k_ln<<<65536, 32, 0, stream>>>(hbuf, ln1_g + i*128, ln1_b + i*128, xb, shift, 1);
    k_btab<<<32768/256, 256, 0, stream>>>(rpb + i*343*8, btab);
    k_gemm<128, EpiQKV><<<dim3(6, 256), blk, 0, stream>>>(
        xb, qkvw + i*49152, EpiQKV{qkv_b + i*384, big});
    k_attn<<<8192, 32, 0, stream>>>(big, btab, attno, shift);
    k_gemm<128, EpiProjRes><<<dim3(2, 256), blk, 0, stream>>>(
        attno, projw + i*16384, EpiProjRes{proj_b + i*128, hbuf, shift});
    k_ln<<<65536, 32, 0, stream>>>(hbuf, ln2_g + i*128, ln2_b + i*128, xb, 0, 0);
    k_gemm<128, EpiFC1Gelu><<<dim3(8, 256), blk, 0, stream>>>(
        xb, fc1w + i*65536, EpiFC1Gelu{fc1_b + i*512, big});
    k_gemm<512, EpiFC2Res><<<dim3(2, 256), blk, 0, stream>>>(
        big, fc2w + i*65536, EpiFC2Res{fc2_b + i*128, hbuf});
  }

  k_dw<<<32768, 256, 0, stream>>>(hbuf, dw_w, xb);
  k_gemm<128, EpiPW><<<dim3(4, 256), blk, 0, stream>>>(
      xb, pww, EpiPW{pw_b, bn_g, bn_b, out});
}
